// AdvancedSparseFocusedAttention_35545149342331
// MI455X (gfx1250) — compile-verified, matrix operand order checked by reading, not run on hardware
//
#include <hip/hip_runtime.h>
#include <hip/hip_bf16.h>

// ---------------------------------------------------------------------------
// AdvancedSparseFocusedAttention for MI455X (gfx1250), wave32 + WMMA.
// B=4, N=4096, D=1024, H=16, HD=64, KEEP=44.
// Round 3: fix async-to-LDS builtin signature (v4i* operands, AS1/AS3).
// ---------------------------------------------------------------------------

typedef __bf16 v16bf __attribute__((ext_vector_type(16)));
typedef __bf16 v8bf  __attribute__((ext_vector_type(8)));
typedef __bf16 v4bf  __attribute__((ext_vector_type(4)));
typedef float  v8f   __attribute__((ext_vector_type(8)));
typedef float  v2f   __attribute__((ext_vector_type(2)));
typedef int    v4i   __attribute__((vector_size(16)));

#define CN   4096
#define CD   1024
#define CH   16
#define CHD  64
#define CBH  64          // B*H
#define CM   16384       // B*N
#define CKEEP 44
#define CEPS 1e-6f

#define GK   1024        // GEMM K (all GEMMs here are 16384x1024x1024)
#define GN   1024
#define GKT  (GK / 32)   // 32 K-stages

#define AS1 __attribute__((address_space(1)))
#define AS3 __attribute__((address_space(3)))

#if defined(__AMDGCN__) && __has_builtin(__builtin_amdgcn_global_load_async_to_lds_b128)
#define HAVE_ASYNC_LDS 1
#else
#define HAVE_ASYNC_LDS 0
#endif

#if HAVE_ASYNC_LDS
#if __has_builtin(__builtin_amdgcn_s_wait_asynccnt)
#define WAIT_ASYNC(n) __builtin_amdgcn_s_wait_asynccnt(n)
#else
#define WAIT_ASYNC(n) asm volatile("s_wait_asynccnt %0" ::"n"(n) : "memory")
#endif
#else
#define WAIT_ASYNC(n) ((void)0)
#endif

// 16-byte global -> LDS copy: async (ASYNCcnt-tracked, no VGPR round trip)
// when the toolchain exposes the gfx1250 builtin, sync fallback otherwise.
__device__ inline void cp16_to_lds(const void* g, void* l) {
#if HAVE_ASYNC_LDS
  __builtin_amdgcn_global_load_async_to_lds_b128((AS1 v4i*)g, (AS3 v4i*)l, 0, 0);
#else
  *(uint4*)l = *(const uint4*)g;
#endif
}

// ---------------------------------------------------------------------------
// fp32 -> bf16 conversion (vectorized x4)
// ---------------------------------------------------------------------------
__global__ __launch_bounds__(256) void cvt_bf16_kernel(const float* __restrict__ X,
                                                       __bf16* __restrict__ Y, int n4) {
  int i = blockIdx.x * 256 + threadIdx.x;
  if (i < n4) {
    float4 v = ((const float4*)X)[i];
    v4bf o;
    o[0] = (__bf16)v.x; o[1] = (__bf16)v.y; o[2] = (__bf16)v.z; o[3] = (__bf16)v.w;
    ((v4bf*)Y)[i] = o;
  }
}

// ---------------------------------------------------------------------------
// bf16 A[Mx1024] row-major x B[1024x1024] row-major (C = A * B^T), fp32 C.
// Tile 128x128x32; 256 threads = 8 waves; wave tile 64x32 = 4x2 WMMA tiles.
// Double-buffered LDS; tile k+1 streamed with GLOBAL_LOAD_ASYNC_TO_LDS_B128
// while tile k feeds v_wmma_f32_16x16x32_bf16.
//
// bf16 16x32 A fragment layout (ISA 7.12.2): lane L holds row m=L%16;
// half=L/16; element e: K = (e/8)*16 + half*8 + (e%8) => two 16B LDS reads.
// ---------------------------------------------------------------------------
__device__ inline v16bf frag16(const __bf16* rowp, int half) {
  union { v16bf v; v8bf h[2]; } u;
  u.h[0] = *(const v8bf*)(rowp + half * 8);
  u.h[1] = *(const v8bf*)(rowp + 16 + half * 8);
  return u.v;
}

__global__ __launch_bounds__(256) void gemm_bf16_nt_kernel(
    const __bf16* __restrict__ A, const __bf16* __restrict__ B,
    float* __restrict__ C) {
  __shared__ __bf16 As[2][128][32];
  __shared__ __bf16 Bs[2][128][32];

  const int tid  = threadIdx.x;
  const int lane = tid & 31;
  const int w    = tid >> 5;
  const int lm   = lane & 15;
  const int half = lane >> 4;
  const int m0   = blockIdx.y * 128;
  const int n0   = blockIdx.x * 128;
  const int wm   = w >> 2;      // 0..1  (64 rows each)
  const int wn   = w & 3;       // 0..3  (32 cols each)

  // per-thread staging coords: thread loads tile row `row0`, cols [col0, col0+16)
  const int c0   = tid * 2;
  const int row0 = c0 >> 2;
  const int col0 = (c0 & 3) * 8;

  const __bf16* gA = &A[(size_t)(m0 + row0) * GK + col0];
  const __bf16* gB = &B[(size_t)(n0 + row0) * GK + col0];

  v8f acc[4][2] = {};

  auto prefetch = [&](int kt, int buf) {
    const int k0 = kt * 32;
    cp16_to_lds(gA + k0,     &As[buf][row0][col0]);
    cp16_to_lds(gA + k0 + 8, &As[buf][row0][col0 + 8]);
    cp16_to_lds(gB + k0,     &Bs[buf][row0][col0]);
    cp16_to_lds(gB + k0 + 8, &Bs[buf][row0][col0 + 8]);
  };

  auto compute = [&](int cur) {
    v16bf af[4], bfr[2];
#pragma unroll
    for (int i = 0; i < 4; i++) af[i]  = frag16(&As[cur][wm * 64 + i * 16 + lm][0], half);
#pragma unroll
    for (int j = 0; j < 2; j++) bfr[j] = frag16(&Bs[cur][wn * 32 + j * 16 + lm][0], half);
#pragma unroll
    for (int i = 0; i < 4; i++)
#pragma unroll
      for (int j = 0; j < 2; j++)
        acc[i][j] = __builtin_amdgcn_wmma_f32_16x16x32_bf16(
            false, af[i], false, bfr[j], (short)0, acc[i][j], false, false);
  };

  prefetch(0, 0);
  for (int kt = 0; kt < GKT - 1; ++kt) {
    const int cur = kt & 1;
    prefetch(kt + 1, cur ^ 1);   // overlap next stage with this stage's WMMA
    WAIT_ASYNC(4);               // 4 async loads still in flight (next stage)
    __syncthreads();
    compute(cur);
    __syncthreads();             // everyone done reading `cur` before reuse
  }
  WAIT_ASYNC(0);
  __syncthreads();
  compute((GKT - 1) & 1);

  // C/D layout: VGPR r -> row r (lanes 0-15) / row 8+r (lanes 16-31), col=lane%16
  float* cp = C + (size_t)(m0 + wm * 64 + half * 8) * GN + n0 + wn * 32 + lm;
#pragma unroll
  for (int i = 0; i < 4; i++)
#pragma unroll
    for (int j = 0; j < 2; j++)
#pragma unroll
      for (int r = 0; r < 8; r++)
        cp[(size_t)(i * 16 + r) * GN + j * 16] = acc[i][j][r];
}

// ---------------------------------------------------------------------------
// focus(): x = abs(leaky(x)) + eps; y = softmax(log(x)*f) * mean(x)
// One block (256 threads) per row of 1024, 4 contiguous floats per thread.
// ---------------------------------------------------------------------------
__device__ inline float wave_sum32(float v) {
#pragma unroll
  for (int o = 16; o > 0; o >>= 1) v += __shfl_xor(v, o, 32);
  return v;
}
__device__ inline float wave_max32(float v) {
#pragma unroll
  for (int o = 16; o > 0; o >>= 1) v = fmaxf(v, __shfl_xor(v, o, 32));
  return v;
}
__device__ inline float block_sum(float v, float* sred, int tid) {
  int lane = tid & 31, w = tid >> 5;
  v = wave_sum32(v);
  if (lane == 0) sred[w] = v;
  __syncthreads();
  float t = 0.f;
#pragma unroll
  for (int i = 0; i < 8; i++) t += sred[i];
  __syncthreads();
  return t;
}
__device__ inline float block_max(float v, float* sred, int tid) {
  int lane = tid & 31, w = tid >> 5;
  v = wave_max32(v);
  if (lane == 0) sred[w] = v;
  __syncthreads();
  float t = -3.4e38f;
#pragma unroll
  for (int i = 0; i < 8; i++) t = fmaxf(t, sred[i]);
  __syncthreads();
  return t;
}

__global__ __launch_bounds__(256) void focus_kernel(const float* __restrict__ X,
                                                    float* __restrict__ Y,
                                                    const float* __restrict__ ff) {
  __shared__ float sred[8];
  const int tid = threadIdx.x;
  const size_t row = blockIdx.x;
  const float f = ff[0];

  float4 v = *(const float4*)&X[row * CD + tid * 4];
  float x[4] = {v.x, v.y, v.z, v.w};
#pragma unroll
  for (int i = 0; i < 4; i++) {
    float t = x[i];
    t = (t >= 0.f) ? t : (-0.01f * t);   // |leaky_relu(x)|
    x[i] = t + CEPS;
  }
  float nrm = block_sum(x[0] + x[1] + x[2] + x[3], sred, tid) * (1.0f / (float)CD);

  float l[4];
#pragma unroll
  for (int i = 0; i < 4; i++) l[i] = __logf(x[i]) * f;
  float mx = block_max(fmaxf(fmaxf(l[0], l[1]), fmaxf(l[2], l[3])), sred, tid);

  float e[4], es = 0.f;
#pragma unroll
  for (int i = 0; i < 4; i++) { e[i] = __expf(l[i] - mx); es += e[i]; }
  float denom = block_sum(es, sred, tid);
  float sc = nrm / denom;

  float4 o = {e[0] * sc, e[1] * sc, e[2] * sc, e[3] * sc};
  *(float4*)&Y[row * CD + tid * 4] = o;
}

// ---------------------------------------------------------------------------
// Head split + top-44-of-64 magnitude sparsify.
// [B,N,D] -> [(B*H), N, HD]; one wave per 64-wide row (2 elems per lane);
// rank by exhaustive comparison with index tie-break (matches lax.top_k).
// ---------------------------------------------------------------------------
__global__ __launch_bounds__(256) void topk_heads_kernel(const float* __restrict__ X,
                                                         float* __restrict__ Yh) {
  __shared__ float ab[8][64];
  const int tid = threadIdx.x, lane = tid & 31, w = tid >> 5;
  const size_t r = (size_t)blockIdx.x * 8 + w;  // 0 .. BH*N-1
  const int bh = (int)(r >> 12);
  const int n  = (int)(r & (CN - 1));
  const int b  = bh >> 4, h = bh & 15;

  const float* src = &X[((size_t)(b * CN + n)) * CD + h * CHD];
  float v0 = src[2 * lane], v1 = src[2 * lane + 1];
  float a0 = fabsf(v0), a1 = fabsf(v1);
  ab[w][2 * lane] = a0;
  ab[w][2 * lane + 1] = a1;
  __syncthreads();

  int c0 = 0, c1 = 0;
#pragma unroll
  for (int j = 0; j < 64; j++) {
    float aj = ab[w][j];
    c0 += (aj > a0) || (aj == a0 && j < 2 * lane);
    c1 += (aj > a1) || (aj == a1 && j < 2 * lane + 1);
  }
  float* dst = &Yh[((size_t)bh * CN + n) * CHD];
  dst[2 * lane]     = (c0 < CKEEP) ? v0 : 0.f;
  dst[2 * lane + 1] = (c1 < CKEEP) ? v1 : 0.f;
}

// ---------------------------------------------------------------------------
// k_mean[bh, d] = mean_n kh[bh, n, d]
// ---------------------------------------------------------------------------
__global__ __launch_bounds__(256) void kmean_kernel(const float* __restrict__ KH,
                                                    float* __restrict__ KM) {
  __shared__ float part[4][64];
  const int bh = blockIdx.x;
  const int tid = threadIdx.x, d = tid & 63, seg = tid >> 6;
  float s = 0.f;
  for (int n = seg; n < CN; n += 4) s += KH[((size_t)bh * CN + n) * CHD + d];
  part[seg][d] = s;
  __syncthreads();
  if (tid < 64) {
    float t = part[0][tid] + part[1][tid] + part[2][tid] + part[3][tid];
    KM[bh * CHD + tid] = t * (1.0f / (float)CN);
  }
}

// ---------------------------------------------------------------------------
// z[bh, n] = qh[bh,n,:] . k_mean[bh,:] + eps
// ---------------------------------------------------------------------------
__global__ __launch_bounds__(256) void zinv_kernel(const float* __restrict__ QH,
                                                   const float* __restrict__ KM,
                                                   float* __restrict__ Z) {
  __shared__ float km[64];
  const int bh = blockIdx.y;
  const int n = blockIdx.x * 256 + threadIdx.x;
  if (threadIdx.x < 64) km[threadIdx.x] = KM[bh * CHD + threadIdx.x];
  __syncthreads();
  const float* q = &QH[((size_t)bh * CN + n) * CHD];
  float s = 0.f;
#pragma unroll
  for (int c = 0; c < CHD; c++) s += q[c] * km[c];
  Z[(size_t)bh * CN + n] = s + CEPS;
}

// ---------------------------------------------------------------------------
// kv[bh] = (1/N) * kh[bh]^T . vh[bh]   (64x64, K=4096)  via V_WMMA_F32_16X16X4
// vh read directly from projected v [B,N,D] with head offset.
// One block per bh; async-staged 64-row chunks; wave -> 1x2 of the 4x4 tiles.
// f32 A 16x4 fragment: elem e -> K = 2*(lane/16) + e, row m = lane%16.
// ---------------------------------------------------------------------------
__global__ __launch_bounds__(256) void kv_kernel(const float* __restrict__ KH,
                                                 const float* __restrict__ VP,
                                                 float* __restrict__ KV) {
  __shared__ float Ks[64][68];
  __shared__ float Vs[64][68];
  const int bh = blockIdx.x;
  const int b = bh >> 4, h = bh & 15;
  const int tid = threadIdx.x, lane = tid & 31, w = tid >> 5;
  const int lm = lane & 15, half = lane >> 4;
  const int ci = w >> 1;            // 0..3 : output row tile
  const int di0 = (w & 1) * 2;      // 0 or 2: output col tile pair

  v8f acc[2] = {};

  for (int n0 = 0; n0 < CN; n0 += 64) {
#pragma unroll
    for (int c = tid; c < 1024; c += 256) {
      int row = c >> 4, col = (c & 15) * 4;
      cp16_to_lds(&KH[((size_t)bh * CN + n0 + row) * CHD + col], &Ks[row][col]);
      cp16_to_lds(&VP[((size_t)(b * CN + n0 + row)) * CD + h * CHD + col], &Vs[row][col]);
    }
    WAIT_ASYNC(0);
    __syncthreads();

#pragma unroll
    for (int kk = 0; kk < 16; kk++) {
      int nb = kk * 4 + half * 2;
      v2f a;  a[0]  = Ks[nb][ci * 16 + lm];        a[1]  = Ks[nb + 1][ci * 16 + lm];
      v2f b0; b0[0] = Vs[nb][di0 * 16 + lm];       b0[1] = Vs[nb + 1][di0 * 16 + lm];
      v2f b1; b1[0] = Vs[nb][di0 * 16 + 16 + lm];  b1[1] = Vs[nb + 1][di0 * 16 + 16 + lm];
      acc[0] = __builtin_amdgcn_wmma_f32_16x16x4_f32(false, a, false, b0, (short)0,
                                                     acc[0], false, false);
      acc[1] = __builtin_amdgcn_wmma_f32_16x16x4_f32(false, a, false, b1, (short)0,
                                                     acc[1], false, false);
    }
    __syncthreads();
  }

  const float s = 1.0f / (float)CN;
#pragma unroll
  for (int j = 0; j < 2; j++)
#pragma unroll
    for (int r = 0; r < 8; r++) {
      int row = ci * 16 + r + half * 8;
      int col = (di0 + j) * 16 + lm;
      KV[(size_t)bh * 4096 + row * 64 + col] = acc[j][r] * s;
    }
}

// ---------------------------------------------------------------------------
// y[bh, n, d] = (qh[bh,n,:] . kv[bh,:,d]) / z[bh,n], written un-head-transposed
// as bf16 [B,N,D] feeding the final GEMM. Tile: 128 n-rows x 64 d per block.
// ---------------------------------------------------------------------------
__global__ __launch_bounds__(256) void y_kernel(const float* __restrict__ QH,
                                                const float* __restrict__ KVm,
                                                const float* __restrict__ Z,
                                                __bf16* __restrict__ Ybf) {
  __shared__ float kvs[64][68];
  const int bh = blockIdx.y;
  const int n0 = blockIdx.x * 128;
  const int b = bh >> 4, h = bh & 15;
  const int tid = threadIdx.x, lane = tid & 31, w = tid >> 5;
  const int lm = lane & 15, half = lane >> 4;
  const int wm = w >> 1;          // 0..3 : 32-row slice
  const int wn = w & 1;           // 0..1 : 32-col slice

#pragma unroll
  for (int c = tid; c < 1024; c += 256) {
    int row = c >> 4, col = (c & 15) * 4;
    cp16_to_lds(&KVm[(size_t)bh * 4096 + row * 64 + col], &kvs[row][col]);
  }
  WAIT_ASYNC(0);
  __syncthreads();

  v8f acc[2][2] = {};
#pragma unroll
  for (int kk = 0; kk < 16; kk++) {
    int kb = kk * 4 + half * 2;
    v2f a[2];
#pragma unroll
    for (int i = 0; i < 2; i++) {
      int n = n0 + wm * 32 + i * 16 + lm;
      const float2 p = *(const float2*)&QH[((size_t)bh * CN + n) * CHD + kb];
      a[i][0] = p.x; a[i][1] = p.y;
    }
    v2f bb[2];
#pragma unroll
    for (int j = 0; j < 2; j++) {
      bb[j][0] = kvs[kb][wn * 32 + j * 16 + lm];
      bb[j][1] = kvs[kb + 1][wn * 32 + j * 16 + lm];
    }
#pragma unroll
    for (int i = 0; i < 2; i++)
#pragma unroll
      for (int j = 0; j < 2; j++)
        acc[i][j] = __builtin_amdgcn_wmma_f32_16x16x4_f32(false, a[i], false, bb[j],
                                                          (short)0, acc[i][j], false, false);
  }

#pragma unroll
  for (int i = 0; i < 2; i++)
#pragma unroll
    for (int j = 0; j < 2; j++)
#pragma unroll
      for (int r = 0; r < 8; r++) {
        int n = n0 + wm * 32 + i * 16 + r + half * 8;
        int d = wn * 32 + j * 16 + lm;
        float zv = Z[(size_t)bh * CN + n];
        float val = acc[i][j][r] / zv;
        Ybf[((size_t)(b * CN + n)) * CD + h * CHD + d] = (__bf16)val;
      }
}

// ---------------------------------------------------------------------------
// Launch
// ---------------------------------------------------------------------------
extern "C" void kernel_launch(void* const* d_in, const int* in_sizes, int n_in,
                              void* d_out, int out_size, void* d_ws, size_t ws_size,
                              hipStream_t stream) {
  const float* q  = (const float*)d_in[0];
  const float* k  = (const float*)d_in[1];
  const float* v  = (const float*)d_in[2];
  const float* Wq = (const float*)d_in[3];
  const float* Wk = (const float*)d_in[4];
  const float* Wv = (const float*)d_in[5];
  const float* Wp = (const float*)d_in[6];
  const float* ff = (const float*)d_in[7];
  float* out = (float*)d_out;

  char* ws = (char*)d_ws;
  __bf16* xbf = (__bf16*)(ws + 0);                 // 33.55 MB activation bf16 scratch
  __bf16* wqb = (__bf16*)(ws + 33554432);          // 4x2MB weights bf16
  __bf16* wkb = wqb + 1048576;
  __bf16* wvb = wkb + 1048576;
  __bf16* wpb = wvb + 1048576;
  float*  qp  = (float*)(ws + 41943040);           // 67.1 MB each
  float*  kp  = (float*)(ws + 109051904);
  float*  vp  = (float*)(ws + 176160768);
  float*  qh  = (float*)(ws + 243269632);
  float*  kh  = (float*)(ws + 310378496);
  float*  km  = (float*)(ws + 377487360);          // 16 KB
  float*  z   = (float*)(ws + 377503744);          // 1 MB
  float*  kv  = (float*)(ws + 378552320);          // 1 MB
  __bf16* ybf = (__bf16*)qp;                       // qp dead after topk -> reuse

  const int NE = CM * CD;        // 16,777,216
  const int WE = CD * CD;        // 1,048,576

  // weights -> bf16
  cvt_bf16_kernel<<<WE / 1024, 256, 0, stream>>>(Wq, wqb, WE / 4);
  cvt_bf16_kernel<<<WE / 1024, 256, 0, stream>>>(Wk, wkb, WE / 4);
  cvt_bf16_kernel<<<WE / 1024, 256, 0, stream>>>(Wv, wvb, WE / 4);
  cvt_bf16_kernel<<<WE / 1024, 256, 0, stream>>>(Wp, wpb, WE / 4);

  dim3 ggrid(CD / 128, CM / 128);   // (8, 128)

  // projections: X * W^T
  cvt_bf16_kernel<<<NE / 1024, 256, 0, stream>>>(q, xbf, NE / 4);
  gemm_bf16_nt_kernel<<<ggrid, 256, 0, stream>>>(xbf, wqb, qp);
  cvt_bf16_kernel<<<NE / 1024, 256, 0, stream>>>(k, xbf, NE / 4);
  gemm_bf16_nt_kernel<<<ggrid, 256, 0, stream>>>(xbf, wkb, kp);
  cvt_bf16_kernel<<<NE / 1024, 256, 0, stream>>>(v, xbf, NE / 4);
  gemm_bf16_nt_kernel<<<ggrid, 256, 0, stream>>>(xbf, wvb, vp);

  // focusing (in-place) then head-split + top-k sparsify
  focus_kernel<<<CM, 256, 0, stream>>>(qp, qp, ff);
  focus_kernel<<<CM, 256, 0, stream>>>(kp, kp, ff);
  topk_heads_kernel<<<(CBH * CN) / 8, 256, 0, stream>>>(qp, qh);
  topk_heads_kernel<<<(CBH * CN) / 8, 256, 0, stream>>>(kp, kh);

  // linear-attention core
  kmean_kernel<<<CBH, 256, 0, stream>>>(kh, km);
  zinv_kernel<<<dim3(CN / 256, CBH), 256, 0, stream>>>(qh, km, z);
  kv_kernel<<<CBH, 256, 0, stream>>>(kh, vp, kv);
  y_kernel<<<dim3(CN / 128, CBH), 256, 0, stream>>>(qh, kv, z, ybf);

  // output projection: y * Wp^T
  gemm_bf16_nt_kernel<<<ggrid, 256, 0, stream>>>(ybf, wpb, out);
}
